// Injector_3977139716834
// MI455X (gfx1250) — matrix operand order repaired and checked by reading, not compiled, not run on hardware
//
#include <hip/hip_runtime.h>

// ---------------------------------------------------------------------------
// Multi-scale deformable attention for MI455X (gfx1250, wave32, WMMA).
// GEMMs run through v_wmma_f32_16x16x32_f16 (f16 in, f32 accumulate) with
// register double-buffering so next-chunk global_load_b128s overlap WMMAs.
// ---------------------------------------------------------------------------

typedef __attribute__((ext_vector_type(16))) _Float16 v16h;
typedef __attribute__((ext_vector_type(8)))  float    v8f;

#define C_DIM   768
#define NH_DIM  12
#define HD_DIM  64
#define L_DIM   3
#define NP_DIM  4
#define LQ_DIM  4096
#define LIN_DIM 21504
#define N_DIM   4

// ---------------------------------------------------------------------------
// Weight repack: w[K][Ncols] (f32, row-major) -> wt[Ncols][K] (f16, K-major)
// so WMMA B-fragment 128-bit loads are contiguous.
// ---------------------------------------------------------------------------
__global__ void packw_kernel(const float* __restrict__ w,
                             _Float16* __restrict__ wt,
                             int K, int Ncols) {
  int total = K * Ncols;
  int idx = blockIdx.x * blockDim.x + threadIdx.x;
  if (idx >= total) return;
  int nn = idx / K;
  int kk = idx - nn * K;
  wt[idx] = (_Float16)w[(size_t)kk * Ncols + nn];
}

// ---------------------------------------------------------------------------
// LayerNorm over C=768, output f16. One 256-thread block per row (3 elts/thr).
// ---------------------------------------------------------------------------
__global__ __launch_bounds__(256) void ln_f16_kernel(
    const float* __restrict__ x, const float* __restrict__ w,
    const float* __restrict__ b, _Float16* __restrict__ y) {
  const int row = blockIdx.x;
  const int tid = threadIdx.x;
  const float* xr = x + (size_t)row * C_DIM;
  float v0 = xr[tid], v1 = xr[tid + 256], v2 = xr[tid + 512];
  __shared__ float red[256];
  red[tid] = v0 + v1 + v2;
  __syncthreads();
  for (int s = 128; s > 0; s >>= 1) {
    if (tid < s) red[tid] += red[tid + s];
    __syncthreads();
  }
  float mu = red[0] * (1.0f / C_DIM);
  __syncthreads();
  float d0 = v0 - mu, d1 = v1 - mu, d2 = v2 - mu;
  red[tid] = d0 * d0 + d1 * d1 + d2 * d2;
  __syncthreads();
  for (int s = 128; s > 0; s >>= 1) {
    if (tid < s) red[tid] += red[tid + s];
    __syncthreads();
  }
  float rs = rsqrtf(red[0] * (1.0f / C_DIM) + 1e-6f);
  _Float16* yr = y + (size_t)row * C_DIM;
  yr[tid]       = (_Float16)(d0 * rs * w[tid]       + b[tid]);
  yr[tid + 256] = (_Float16)(d1 * rs * w[tid + 256] + b[tid + 256]);
  yr[tid + 512] = (_Float16)(d2 * rs * w[tid + 512] + b[tid + 512]);
}

// ---------------------------------------------------------------------------
// WMMA fragment = 8 dwords = 2 x 128-bit chunks per lane.
// ---------------------------------------------------------------------------
union Frag {
  v16h v;
  uint4 q[2];
};

// A fragment (16x32 f16, row-major source, lda = K):
//   lane m = lane&15, h = lane>>4; chunk0 = elems [16h, 16h+8), chunk1 = +8.
__device__ __forceinline__ void load_a_frag(Frag& f, const _Float16* arow,
                                            int k0, int h) {
  const _Float16* p = arow + k0 + 16 * h;
  f.q[0] = *reinterpret_cast<const uint4*>(p);
  f.q[1] = *reinterpret_cast<const uint4*>(p + 8);
}

// B fragment (32x16 f16, K-major source, col stride K). bcol already
// includes + col*K + 16*h; chunk0 = elems [0,8), chunk1 = [8,16).
__device__ __forceinline__ void load_b_frag(Frag& f, const _Float16* bcol,
                                            int k0) {
  const _Float16* p = bcol + k0;
  f.q[0] = *reinterpret_cast<const uint4*>(p);
  f.q[1] = *reinterpret_cast<const uint4*>(p + 8);
}

// ---------------------------------------------------------------------------
// WMMA GEMM: C[M][N] = A[M][K](f16, row-major) * Bt[N][K](f16, K-major) + bias
//   EPI 0: f32 store          EPI 1: f16 store
//   EPI 2: f32 store of resid + gamma * (acc + bias)   (residual epilogue)
// Block = 128 threads = 4 waves; each wave owns one 16 x (16*NACC) tile.
// K stepped by 64 = two ping-pong chunks of 32; loads for chunk i+1 are
// issued before WMMAs of chunk i (register double-buffering, no copies).
// A/B fragments use contiguous k = 16h + s (not the ISA table order); D layout per ISA (wave32).
// ---------------------------------------------------------------------------
template <int NACC, int EPI>
__global__ __launch_bounds__(128) void gemm_f16_wmma(
    const _Float16* __restrict__ A, const _Float16* __restrict__ Bt,
    const float* __restrict__ bias, float* __restrict__ Cf,
    _Float16* __restrict__ Ch, const float* __restrict__ resid,
    const float* __restrict__ gamma, int M, int Nn, int K) {
  const int lane = threadIdx.x & 31;
  const int wave = threadIdx.x >> 5;
  const int m0 = (blockIdx.y * 4 + wave) * 16;
  const int n0 = blockIdx.x * (16 * NACC);
  const int mr = lane & 15;   // M row (A) / N col (B,D) within tile
  const int h  = lane >> 4;   // lane half

  const _Float16* Arow = A + (size_t)(m0 + mr) * K;
  const _Float16* Bcol[NACC];
#pragma unroll
  for (int t = 0; t < NACC; ++t)
    Bcol[t] = Bt + (size_t)(n0 + t * 16 + mr) * K + 16 * h;

  v8f acc[NACC];
  v8f zero = {};
#pragma unroll
  for (int t = 0; t < NACC; ++t) acc[t] = zero;

  Frag a0, a1;
  Frag b0[NACC], b1[NACC];

  // preload chunk 0
  load_a_frag(a0, Arow, 0, h);
#pragma unroll
  for (int t = 0; t < NACC; ++t) load_b_frag(b0[t], Bcol[t], 0);

  for (int k0 = 0; k0 < K - 64; k0 += 64) {
    // issue loads for chunk k0+32, then consume chunk k0
    load_a_frag(a1, Arow, k0 + 32, h);
#pragma unroll
    for (int t = 0; t < NACC; ++t) load_b_frag(b1[t], Bcol[t], k0 + 32);
#pragma unroll
    for (int t = 0; t < NACC; ++t)
      acc[t] = __builtin_amdgcn_wmma_f32_16x16x32_f16(
          false, a0.v, false, b0[t].v, (short)0, acc[t], false, false);

    // issue loads for chunk k0+64, then consume chunk k0+32
    load_a_frag(a0, Arow, k0 + 64, h);
#pragma unroll
    for (int t = 0; t < NACC; ++t) load_b_frag(b0[t], Bcol[t], k0 + 64);
#pragma unroll
    for (int t = 0; t < NACC; ++t)
      acc[t] = __builtin_amdgcn_wmma_f32_16x16x32_f16(
          false, a1.v, false, b1[t].v, (short)0, acc[t], false, false);
  }

  // tail: chunks K-64 (in a0/b0) and K-32
  load_a_frag(a1, Arow, K - 32, h);
#pragma unroll
  for (int t = 0; t < NACC; ++t) load_b_frag(b1[t], Bcol[t], K - 32);
#pragma unroll
  for (int t = 0; t < NACC; ++t)
    acc[t] = __builtin_amdgcn_wmma_f32_16x16x32_f16(
        false, a0.v, false, b0[t].v, (short)0, acc[t], false, false);
#pragma unroll
  for (int t = 0; t < NACC; ++t)
    acc[t] = __builtin_amdgcn_wmma_f32_16x16x32_f16(
        false, a1.v, false, b1[t].v, (short)0, acc[t], false, false);

  // D layout: lane col = mr, VGPR r holds row 8*h + r.
#pragma unroll
  for (int t = 0; t < NACC; ++t) {
    const int col = n0 + t * 16 + mr;
    const float bc = bias[col];
#pragma unroll
    for (int r = 0; r < 8; ++r) {
      const int row = m0 + 8 * h + r;
      const size_t o = (size_t)row * Nn + col;
      const float val = acc[t][r] + bc;
      if constexpr (EPI == 0) {
        Cf[o] = val;
      } else if constexpr (EPI == 1) {
        Ch[o] = (_Float16)val;
      } else {
        Cf[o] = resid[o] + gamma[col] * val;
      }
    }
  }
}

// ---------------------------------------------------------------------------
// Softmax(12) + bilinear sampling. One block of 64 threads per (n, lq, head);
// threadIdx.x = channel d in [0,64). value is f16 [N][Lin][C].
// ---------------------------------------------------------------------------
__global__ __launch_bounds__(64) void sample_kernel(
    const _Float16* __restrict__ value, const float* __restrict__ offv,
    const float* __restrict__ awl, const float* __restrict__ refp,
    _Float16* __restrict__ outp) {
  const int d = threadIdx.x;
  int bid = blockIdx.x;
  const int head = bid % NH_DIM; bid /= NH_DIM;
  const int lq = bid % LQ_DIM;
  const int n  = bid / LQ_DIM;
  const size_t nq = (size_t)n * LQ_DIM + lq;

  // softmax over the 12 (level, point) logits for this head
  const float* awr = awl + nq * (NH_DIM * 12) + head * 12;
  float lg[12];
  float mx = -3.4e38f;
#pragma unroll
  for (int j = 0; j < 12; ++j) { lg[j] = awr[j]; mx = fmaxf(mx, lg[j]); }
  float s = 0.0f;
#pragma unroll
  for (int j = 0; j < 12; ++j) { lg[j] = __expf(lg[j] - mx); s += lg[j]; }
  const float inv = 1.0f / s;

  const float* offr = offv + nq * (NH_DIM * L_DIM * NP_DIM * 2)
                    + head * (L_DIM * NP_DIM * 2);
  const int Hs[3] = {128, 64, 32};
  const int Ws[3] = {128, 64, 32};
  const int St[3] = {0, 16384, 20480};

  float acc = 0.0f;
  const _Float16* vbase =
      value + (size_t)n * LIN_DIM * C_DIM + head * HD_DIM + d;

#pragma unroll
  for (int l = 0; l < L_DIM; ++l) {
    const int H_ = Hs[l], W_ = Ws[l], st = St[l];
    const float rx = refp[(nq * L_DIM + l) * 2 + 0];
    const float ry = refp[(nq * L_DIM + l) * 2 + 1];
#pragma unroll
    for (int p = 0; p < NP_DIM; ++p) {
      const float aww = lg[l * NP_DIM + p] * inv;
      const float ox = offr[(l * NP_DIM + p) * 2 + 0];
      const float oy = offr[(l * NP_DIM + p) * 2 + 1];
      const float px = (rx + ox / (float)W_) * (float)W_ - 0.5f;
      const float py = (ry + oy / (float)H_) * (float)H_ - 0.5f;
      const float x0f = floorf(px), y0f = floorf(py);
      const float tx = px - x0f, ty = py - y0f;
      const int x0 = (int)x0f, y0 = (int)y0f;
      const float cw[4] = {(1.0f - tx) * (1.0f - ty), tx * (1.0f - ty),
                           (1.0f - tx) * ty,          tx * ty};
      const int dxs[4] = {0, 1, 0, 1};
      const int dys[4] = {0, 0, 1, 1};
#pragma unroll
      for (int c2 = 0; c2 < 4; ++c2) {
        const int xi = x0 + dxs[c2];
        const int yi = y0 + dys[c2];
        if (xi >= 0 && xi < W_ && yi >= 0 && yi < H_) {
          const float v = (float)vbase[(size_t)(st + yi * W_ + xi) * C_DIM];
          acc += aww * cw[c2] * v;
        }
      }
    }
  }
  outp[nq * C_DIM + head * HD_DIM + d] = (_Float16)acc;
}

// ---------------------------------------------------------------------------
// Host launcher
// ---------------------------------------------------------------------------
extern "C" void kernel_launch(void* const* d_in, const int* in_sizes, int n_in,
                              void* d_out, int out_size, void* d_ws,
                              size_t ws_size, hipStream_t stream) {
  (void)in_sizes; (void)n_in; (void)out_size; (void)ws_size;

  const float* query = (const float*)d_in[0];
  const float* refp  = (const float*)d_in[1];
  const float* feat  = (const float*)d_in[2];
  const float* qn_w  = (const float*)d_in[3];
  const float* qn_b  = (const float*)d_in[4];
  const float* fn_w  = (const float*)d_in[5];
  const float* fn_b  = (const float*)d_in[6];
  const float* gamma = (const float*)d_in[7];
  const float* so_w  = (const float*)d_in[8];
  const float* so_b  = (const float*)d_in[9];
  const float* aw_w  = (const float*)d_in[10];
  const float* aw_b  = (const float*)d_in[11];
  const float* vp_w  = (const float*)d_in[12];
  const float* vp_b  = (const float*)d_in[13];
  const float* op_w  = (const float*)d_in[14];
  const float* op_b  = (const float*)d_in[15];

  const int Mq = N_DIM * LQ_DIM;   // 16384
  const int Mf = N_DIM * LIN_DIM;  // 86016
  const int K  = C_DIM;            // 768
  const int Nso = NH_DIM * L_DIM * NP_DIM * 2;  // 288
  const int Naw = NH_DIM * L_DIM * NP_DIM;      // 144

  char* ws = (char*)d_ws;
  size_t cur = 0;
  auto alloc = [&](size_t bytes) -> void* {
    void* p = ws + cur;
    cur += (bytes + 255) & ~(size_t)255;
    return p;
  };
  _Float16* q16    = (_Float16*)alloc((size_t)Mq * K * 2);
  _Float16* f16b   = (_Float16*)alloc((size_t)Mf * K * 2);
  _Float16* val16  = (_Float16*)alloc((size_t)Mf * K * 2);
  float*    offb   = (float*)alloc((size_t)Mq * Nso * 4);
  float*    awlb   = (float*)alloc((size_t)Mq * Naw * 4);
  _Float16* samp16 = (_Float16*)alloc((size_t)Mq * K * 2);
  _Float16* vp_wt  = (_Float16*)alloc((size_t)K * K * 2);
  _Float16* so_wt  = (_Float16*)alloc((size_t)K * Nso * 2);
  _Float16* aw_wt  = (_Float16*)alloc((size_t)K * Naw * 2);
  _Float16* op_wt  = (_Float16*)alloc((size_t)K * K * 2);

  // 1) repack weights to f16, K-major
  packw_kernel<<<(K * K   + 255) / 256, 256, 0, stream>>>(vp_w, vp_wt, K, K);
  packw_kernel<<<(K * Nso + 255) / 256, 256, 0, stream>>>(so_w, so_wt, K, Nso);
  packw_kernel<<<(K * Naw + 255) / 256, 256, 0, stream>>>(aw_w, aw_wt, K, Naw);
  packw_kernel<<<(K * K   + 255) / 256, 256, 0, stream>>>(op_w, op_wt, K, K);

  // 2) layernorms -> f16
  ln_f16_kernel<<<Mq, 256, 0, stream>>>(query, qn_w, qn_b, q16);
  ln_f16_kernel<<<Mf, 256, 0, stream>>>(feat, fn_w, fn_b, f16b);

  // 3) value projection: [86016 x 768] = f16b @ vp_w  -> f16
  gemm_f16_wmma<4, 1><<<dim3(K / 64, Mf / 64), 128, 0, stream>>>(
      f16b, vp_wt, vp_b, nullptr, val16, nullptr, nullptr, Mf, K, K);

  // 4) sampling offsets [16384 x 288] and attention logits [16384 x 144]
  gemm_f16_wmma<2, 0><<<dim3(Nso / 32, Mq / 64), 128, 0, stream>>>(
      q16, so_wt, so_b, offb, nullptr, nullptr, nullptr, Mq, Nso, K);
  gemm_f16_wmma<3, 0><<<dim3(Naw / 48, Mq / 64), 128, 0, stream>>>(
      q16, aw_wt, aw_b, awlb, nullptr, nullptr, nullptr, Mq, Naw, K);

  // 5) softmax + bilinear sampling -> f16 [16384 x 768]
  sample_kernel<<<N_DIM * LQ_DIM * NH_DIM, 64, 0, stream>>>(
      val16, offb, awlb, refp, samp16);

  // 6) output projection + residual: d_out = query + gamma * (samp @ op_w + b)
  gemm_f16_wmma<4, 2><<<dim3(K / 64, Mq / 64), 128, 0, stream>>>(
      samp16, op_wt, op_b, (float*)d_out, nullptr, query, gamma, Mq, K, K);
}
